// ReservoirLocalConnectivityBatchSeq_78872779423774
// MI455X (gfx1250) — matrix operand order repaired and checked by reading, not compiled, not run on hardware
//
#include <hip/hip_runtime.h>
#include <hip/hip_bf16.h>

#define I_DIM 256
#define H_DIM 3840
#define B_DIM 32
#define T_DIM 512
#define NWG   120                               // persistent grid (H/32 col strips)

typedef __attribute__((ext_vector_type(16))) __bf16      v16bf;
typedef __attribute__((ext_vector_type(8)))  float        v8f;
typedef __attribute__((ext_vector_type(4)))  unsigned int u32x4;

union Frag16 {                 // 16 bf16 = 32 bytes, loaded as 2x b128
    v16bf          v;
    u32x4          q[2];
    unsigned short s[16];
};

__device__ __forceinline__ unsigned short f2bf(float x) {
    unsigned int u = __float_as_uint(x);
    u += 0x7FFFu + ((u >> 16) & 1u);          // round-to-nearest-even
    return (unsigned short)(u >> 16);
}

__device__ __forceinline__ void load_frag(Frag16& f, const unsigned short* row,
                                          int kb, int off0, int off1) {
    f.q[0] = *(const u32x4*)(row + kb + off0);
    f.q[1] = *(const u32x4*)(row + kb + off1);
}

// device-wide sense-reversing barrier (thread 0 of each WG participates)
__device__ __forceinline__ void grid_sync(unsigned int* bar) {
    __syncthreads();
    if (threadIdx.x == 0) {
        unsigned gen = __hip_atomic_load(&bar[1], __ATOMIC_ACQUIRE,
                                         __HIP_MEMORY_SCOPE_AGENT);
        unsigned arrived = __hip_atomic_fetch_add(&bar[0], 1u, __ATOMIC_ACQ_REL,
                                                  __HIP_MEMORY_SCOPE_AGENT);
        if (arrived == NWG - 1u) {
            __hip_atomic_store(&bar[0], 0u, __ATOMIC_RELAXED,
                               __HIP_MEMORY_SCOPE_AGENT);
            __hip_atomic_fetch_add(&bar[1], 1u, __ATOMIC_ACQ_REL,
                                   __HIP_MEMORY_SCOPE_AGENT);
        } else {
            while (__hip_atomic_load(&bar[1], __ATOMIC_ACQUIRE,
                                     __HIP_MEMORY_SCOPE_AGENT) == gen)
                __builtin_amdgcn_s_sleep(2);
        }
    }
    __syncthreads();
}

// ---------------------------------------------------------------- converts
__global__ void __launch_bounds__(256)
cvt_f32_bf16(const float* __restrict__ in, unsigned short* __restrict__ out, int n) {
    int i = blockIdx.x * 256 + threadIdx.x;
    if (i < n) out[i] = f2bf(in[i]);
}

__global__ void init_barrier(unsigned int* bar) { bar[0] = 0u; bar[1] = 0u; }

// ------------------------------------------------- xw = X @ W_ih^T + biases
// X: (B*T, I) bf16.  W: (H, I) bf16.  out: (B*T, H) f32.  2x2 tile per wave.
__global__ void __launch_bounds__(256)
proj_gemm(const unsigned short* __restrict__ Xb,
          const unsigned short* __restrict__ Wb,
          const float* __restrict__ b_ih, const float* __restrict__ b_hh,
          float* __restrict__ out)
{
    const int lane = threadIdx.x & 31;
    const int wid  = blockIdx.x * 8 + (threadIdx.x >> 5);
    const int NP   = H_DIM / 32;                 // 120 npairs
    const int mp   = wid / NP;                   // 0..511
    const int np   = wid % NP;                   // 0..119
    const int half = lane >> 4;
    const int mn   = lane & 15;
    const int aoff0 = 8 * half, aoff1 = 16 + 8 * half;
    const int boff0 = 8 * half, boff1 = 16 + 8 * half;

    const unsigned short* arow0 = Xb + (size_t)(mp * 32 + mn) * I_DIM;
    const unsigned short* arow1 = arow0 + (size_t)16 * I_DIM;
    const unsigned short* brow0 = Wb + (size_t)(np * 32 + mn) * I_DIM;
    const unsigned short* brow1 = brow0 + (size_t)16 * I_DIM;

    v8f c00 = {}, c01 = {}, c10 = {}, c11 = {};
    #pragma unroll
    for (int kb = 0; kb < I_DIM; kb += 32) {
        Frag16 a0, a1, b0, b1;
        load_frag(a0, arow0, kb, aoff0, aoff1);
        load_frag(a1, arow1, kb, aoff0, aoff1);
        load_frag(b0, brow0, kb, boff0, boff1);
        load_frag(b1, brow1, kb, boff0, boff1);
        c00 = __builtin_amdgcn_wmma_f32_16x16x32_bf16(false, a0.v, false, b0.v, (short)0, c00, false, false);
        c01 = __builtin_amdgcn_wmma_f32_16x16x32_bf16(false, a0.v, false, b1.v, (short)0, c01, false, false);
        c10 = __builtin_amdgcn_wmma_f32_16x16x32_bf16(false, a1.v, false, b0.v, (short)0, c10, false, false);
        c11 = __builtin_amdgcn_wmma_f32_16x16x32_bf16(false, a1.v, false, b1.v, (short)0, c11, false, false);
    }
    const int   j0 = np * 32 + mn, j1 = j0 + 16;
    const float bias0 = b_ih[j0] + b_hh[j0];
    const float bias1 = b_ih[j1] + b_hh[j1];
    #pragma unroll
    for (int r = 0; r < 8; ++r) {
        const size_t row0 = (size_t)(mp * 32 + r + 8 * half);
        const size_t row1 = row0 + 16;
        out[row0 * H_DIM + j0] = c00[r] + bias0;
        out[row0 * H_DIM + j1] = c01[r] + bias1;
        out[row1 * H_DIM + j0] = c10[r] + bias0;
        out[row1 * H_DIM + j1] = c11[r] + bias1;
    }
}

// ---------------- persistent recurrence: all T steps in one launch ---------
// h_t = tanh(xw_t + h_{t-1} @ W_hh^T).  Grid = 120 WGs, each owns a 32-col
// strip.  Wave w = K-slice (8 slices of 480); each wave computes the full
// M=32 x 32-col 2x2 register tile; partials reduced through 32 KB LDS, then
// xw-add + tanh, f32 out + bf16 h ping-pong, device-wide barrier per step.
__global__ void __launch_bounds__(256)
rnn_persistent(unsigned short* hb0, unsigned short* hb1,
               const unsigned short* __restrict__ Whh,
               float* __restrict__ out, unsigned int* bar)
{
    __shared__ float red[8][32][32];             // [kslice][m][ncol] = 32 KB

    const int lane   = threadIdx.x & 31;
    const int kslice = threadIdx.x >> 5;         // 0..7, K = 480 each
    const int half   = lane >> 4;
    const int mn     = lane & 15;
    const int aoff0 = 8 * half, aoff1 = 16 + 8 * half;
    const int boff0 = 8 * half, boff1 = 16 + 8 * half;

    const int ncol0 = blockIdx.x * 32;           // this WG's column strip
    const int k0    = kslice * (H_DIM / 8);      // 0,480,...,3360

    const unsigned short* brow0 = Whh + (size_t)(ncol0 + mn) * H_DIM + k0;
    const unsigned short* brow1 = brow0 + (size_t)16 * H_DIM;

    // reduction-phase indices: 1024 elems / 256 threads = 4 each
    const int e0 = threadIdx.x * 4;
    const int rm = e0 >> 5;                      // row 0..31
    const int rn = e0 & 31;                      // col-in-strip
    const int jb = ncol0 + rn;

    #pragma unroll 1
    for (int t = 0; t < T_DIM; ++t) {
        const unsigned short* hin = (t & 1) ? hb1 : hb0;
        unsigned short*       hout = (t & 1) ? hb0 : hb1;
        float* outt = out + (size_t)t * H_DIM;

        const unsigned short* arow0 = hin + (size_t)mn * H_DIM + k0;
        const unsigned short* arow1 = arow0 + (size_t)16 * H_DIM;

        v8f c00 = {}, c01 = {}, c10 = {}, c11 = {};
        #pragma unroll 3
        for (int kb = 0; kb < H_DIM / 8; kb += 32) {         // 15 iterations
            Frag16 a0, a1, b0, b1;
            load_frag(a0, arow0, kb, aoff0, aoff1);
            load_frag(a1, arow1, kb, aoff0, aoff1);
            load_frag(b0, brow0, kb, boff0, boff1);
            load_frag(b1, brow1, kb, boff0, boff1);
            c00 = __builtin_amdgcn_wmma_f32_16x16x32_bf16(false, a0.v, false, b0.v, (short)0, c00, false, false);
            c01 = __builtin_amdgcn_wmma_f32_16x16x32_bf16(false, a0.v, false, b1.v, (short)0, c01, false, false);
            c10 = __builtin_amdgcn_wmma_f32_16x16x32_bf16(false, a1.v, false, b0.v, (short)0, c10, false, false);
            c11 = __builtin_amdgcn_wmma_f32_16x16x32_bf16(false, a1.v, false, b1.v, (short)0, c11, false, false);
        }

        // spill partials: C tile (mi,ni): m = mi*16 + r + 8*half, n = ni*16 + mn
        #pragma unroll
        for (int r = 0; r < 8; ++r) {
            const int m0 = r + 8 * half, m1 = m0 + 16;
            red[kslice][m0][mn]      = c00[r];
            red[kslice][m0][16 + mn] = c01[r];
            red[kslice][m1][mn]      = c10[r];
            red[kslice][m1][16 + mn] = c11[r];
        }
        __syncthreads();

        // reduce 8 K-slices, add xw (resident in out), tanh, f32 + bf16 h
        #pragma unroll
        for (int r = 0; r < 4; ++r) {
            float sum = 0.f;
            #pragma unroll
            for (int s = 0; s < 8; ++s) sum += red[s][rm][rn + r];
            float* p = outt + (size_t)rm * ((size_t)T_DIM * H_DIM) + jb + r;
            const float v = tanhf(sum + *p);
            *p = v;
            hout[(size_t)rm * H_DIM + jb + r] = f2bf(v);
        }

        grid_sync(bar);   // h_t fully published before any WG starts step t+1
    }
}

// ------------------------------------------------- hidden = out[:, T-1, :]
__global__ void __launch_bounds__(256)
copy_hidden(const float* __restrict__ out, float* __restrict__ hid) {
    int i = blockIdx.x * 256 + threadIdx.x;
    if (i < B_DIM * H_DIM) {
        int b = i / H_DIM, j = i % H_DIM;
        hid[i] = out[(size_t)b * T_DIM * H_DIM + (size_t)(T_DIM - 1) * H_DIM + j];
    }
}

extern "C" void kernel_launch(void* const* d_in, const int* in_sizes, int n_in,
                              void* d_out, int out_size, void* d_ws, size_t ws_size,
                              hipStream_t stream) {
    const float* x    = (const float*)d_in[0];   // (B,T,I)
    const float* h0   = (const float*)d_in[1];   // (B,H)
    const float* W_ih = (const float*)d_in[2];   // (H,I)
    const float* W_hh = (const float*)d_in[3];   // (H,H)
    const float* b_ih = (const float*)d_in[4];
    const float* b_hh = (const float*)d_in[5];

    float* out = (float*)d_out;                              // (B,T,H)
    float* hid = out + (size_t)B_DIM * T_DIM * H_DIM;        // (1,B,H)

    // workspace layout (recreated every call; no cached state)
    unsigned short* whh_bf = (unsigned short*)d_ws;                           // H*H
    unsigned short* wih_bf = whh_bf + (size_t)H_DIM * H_DIM;                  // H*I
    unsigned short* x_bf   = wih_bf + (size_t)H_DIM * I_DIM;                  // B*T*I
    unsigned short* hbuf0  = x_bf   + (size_t)B_DIM * T_DIM * I_DIM;         // B*H
    unsigned short* hbuf1  = hbuf0  + (size_t)B_DIM * H_DIM;                 // B*H
    unsigned int*   bar    = (unsigned int*)(hbuf1 + (size_t)B_DIM * H_DIM);

    const int nwhh = H_DIM * H_DIM;
    const int nwih = H_DIM * I_DIM;
    const int nx   = B_DIM * T_DIM * I_DIM;
    const int nh   = B_DIM * H_DIM;
    cvt_f32_bf16<<<(nwhh + 255) / 256, 256, 0, stream>>>(W_hh, whh_bf, nwhh);
    cvt_f32_bf16<<<(nwih + 255) / 256, 256, 0, stream>>>(W_ih, wih_bf, nwih);
    cvt_f32_bf16<<<(nx   + 255) / 256, 256, 0, stream>>>(x,    x_bf,   nx);
    cvt_f32_bf16<<<(nh   + 255) / 256, 256, 0, stream>>>(h0,   hbuf0,  nh);
    init_barrier<<<1, 1, 0, stream>>>(bar);

    // input projection: xw (+ biases) straight into the output buffer
    const int wjobs = (B_DIM * T_DIM / 32) * (H_DIM / 32);   // 61440 wave jobs
    proj_gemm<<<wjobs / 8, 256, 0, stream>>>(x_bf, wih_bf, b_ih, b_hh, out);

    // whole recurrence in a single persistent launch (512 barrier-separated steps)
    rnn_persistent<<<NWG, 256, 0, stream>>>(hbuf0, hbuf1, whh_bf, out, bar);

    copy_hidden<<<(nh + 255) / 256, 256, 0, stream>>>(out, hid);
}
